// FrameTransformer_23287312679634
// MI455X (gfx1250) — matrix operand order repaired and checked by reading, not compiled, not run on hardware
//
#include <hip/hip_runtime.h>
#include <hip/hip_bf16.h>

// ---------------------------------------------------------------------------
// Types for CDNA5 WMMA (wave32, 16x16x32 bf16 -> f32)
// ---------------------------------------------------------------------------
typedef __bf16 v16bf __attribute__((ext_vector_type(16)));
typedef float  v8f   __attribute__((ext_vector_type(8)));

union Frag {
    v16bf v;
    uint4 q[2];
    unsigned short u[16];
};

#define ELEMS   262144      // 512*512
#define HW_B    2
#define HW_C    16
#define HW_H    512
#define HW_W    512
#define HEADS   8
#define HD      64

__device__ __forceinline__ unsigned short f32_to_bf16(float f) {
    unsigned int u = __float_as_uint(f);
    unsigned int r = (u + 0x7FFFu + ((u >> 16) & 1u)) >> 16;
    return (unsigned short)r;
}

// ---------------------------------------------------------------------------
// Elementwise f32 -> bf16 convert
// ---------------------------------------------------------------------------
__global__ void cvt_bf16_kernel(const float* __restrict__ in,
                                unsigned short* __restrict__ out, int n) {
    int idx = blockIdx.x * 256 + threadIdx.x;
    if (idx < n) out[idx] = f32_to_bf16(in[idx]);
}

// ---------------------------------------------------------------------------
// LayerNorm over H axis (per (b,c,w) column), per-channel affine [C,H]
// grid: B*C*(W/128), block: 128
// ---------------------------------------------------------------------------
__global__ void layernorm_kernel(const float* __restrict__ x,
                                 const float* __restrict__ g,
                                 const float* __restrict__ be,
                                 float* __restrict__ out) {
    int bx = blockIdx.x;
    int w = (bx & 3) * 128 + threadIdx.x;
    int bc = bx >> 2;
    int c = bc & (HW_C - 1);
    const float* xp = x + (size_t)bc * ELEMS + w;
    float s = 0.f, ss = 0.f;
    for (int h = 0; h < HW_H; ++h) {
        float v = xp[h * HW_W];
        s += v; ss += v * v;
    }
    float mu = s * (1.f / HW_H);
    float var = ss * (1.f / HW_H) - mu * mu;
    float rinv = rsqrtf(var + 1e-5f);
    float* op = out + (size_t)bc * ELEMS + w;
    for (int h = 0; h < HW_H; ++h) {
        float v = (xp[h * HW_W] - mu) * rinv;
        op[h * HW_W] = v * g[c * HW_H + h] + be[c * HW_H + h];
    }
}

// ---------------------------------------------------------------------------
// Direct 3x3 SAME conv, C=16 in/out, weights staged in LDS
// grid: B*C*H*W/256, block 256
// ---------------------------------------------------------------------------
__global__ __launch_bounds__(256) void conv3x3_kernel(
    const float* __restrict__ in, const float* __restrict__ w,
    float* __restrict__ out) {
    __shared__ float wl[HW_C * HW_C * 9];
    for (int i = threadIdx.x; i < HW_C * HW_C * 9; i += 256) wl[i] = w[i];
    __syncthreads();
    int idx = blockIdx.x * 256 + threadIdx.x;
    int x = idx & 511;
    int y = (idx >> 9) & 511;
    int co = (idx >> 18) & 15;
    int b = idx >> 22;
    const float* ib = in + (size_t)b * HW_C * ELEMS;
    float acc = 0.f;
    for (int ci = 0; ci < HW_C; ++ci) {
        const float* ip = ib + (size_t)ci * ELEMS;
        const float* wp = &wl[(co * HW_C + ci) * 9];
        for (int dy = -1; dy <= 1; ++dy) {
            int yy = y + dy;
            if ((unsigned)yy >= 512u) continue;
            for (int dx = -1; dx <= 1; ++dx) {
                int xx = x + dx;
                if ((unsigned)xx >= 512u) continue;
                acc += wp[(dy + 1) * 3 + (dx + 1)] * ip[yy * 512 + xx];
            }
        }
    }
    out[idx] = acc;
}

// ---------------------------------------------------------------------------
// Tiled transpose + f32->bf16: in[inst][R=512][C=512] -> out[inst][C][R]
// grid: (16,16,inst), block (32,8)
// ---------------------------------------------------------------------------
__global__ void transpose_cvt_kernel(const float* __restrict__ in,
                                     unsigned short* __restrict__ out) {
    __shared__ float t[32][33];
    int inst = blockIdx.z;
    const float* ip = in + (size_t)inst * ELEMS;
    unsigned short* op = out + (size_t)inst * ELEMS;
    int r0 = blockIdx.y * 32, c0 = blockIdx.x * 32;
    for (int i = 0; i < 4; ++i) {
        int r = threadIdx.y + i * 8;
        t[r][threadIdx.x] = ip[(size_t)(r0 + r) * 512 + c0 + threadIdx.x];
    }
    __syncthreads();
    for (int i = 0; i < 4; ++i) {
        int r = threadIdx.y + i * 8;
        op[(size_t)(c0 + r) * 512 + r0 + threadIdx.x] = f32_to_bf16(t[threadIdx.x][r]);
    }
}

// ---------------------------------------------------------------------------
// Batched bf16 GEMM with f32 accumulate (WMMA 16x16x32 bf16).
//   C[inst][o][w] = sum_i A[inst % numAInst][o][i] * BT[inst][w][i]  (+ res)
// A row-major [M=512,K=512]; BT row-major [N=512,K=512].
// grid: (N/64, M/128, numInst), block 256 (8 waves; wave tile 32x32)
// ---------------------------------------------------------------------------
__global__ __launch_bounds__(256) void gemm_bf16_kernel(
    const unsigned short* __restrict__ A, int numAInst,
    const unsigned short* __restrict__ BT,
    const float* __restrict__ res, float* __restrict__ C) {
    int inst = blockIdx.z;
    const unsigned short* Ab = A + (size_t)(inst % numAInst) * ELEMS;
    const unsigned short* Bb = BT + (size_t)inst * ELEMS;
    float* Cb = C + (size_t)inst * ELEMS;
    const float* Rb = res ? (res + (size_t)inst * ELEMS) : nullptr;

    int tid = threadIdx.x;
    int lane = tid & 31, wid = tid >> 5;
    int wm = wid & 3, wn = wid >> 2;
    int m0 = blockIdx.y * 128 + wm * 32;
    int n0 = blockIdx.x * 64 + wn * 32;
    int ln = lane & 15, hi = lane >> 4;

    v8f acc[2][2] = {};
    for (int kk = 0; kk < 512; kk += 32) {
        Frag a[2], b[2];
#pragma unroll
        for (int t = 0; t < 2; ++t) {
            const unsigned short* ap = Ab + (size_t)(m0 + t * 16 + ln) * 512 + kk + hi * 8;
            a[t].q[0] = *(const uint4*)ap;
            a[t].q[1] = *(const uint4*)(ap + 16);
            __builtin_prefetch(ap + 32, 0, 1);
        }
#pragma unroll
        for (int t = 0; t < 2; ++t) {
            const unsigned short* bp = Bb + (size_t)(n0 + t * 16 + ln) * 512 + kk + hi * 8;
            b[t].q[0] = *(const uint4*)bp;
            b[t].q[1] = *(const uint4*)(bp + 16);
            __builtin_prefetch(bp + 32, 0, 1);
        }
#pragma unroll
        for (int i = 0; i < 2; ++i)
#pragma unroll
            for (int j = 0; j < 2; ++j)
                acc[i][j] = __builtin_amdgcn_wmma_f32_16x16x32_bf16(
                    false, a[i].v, false, b[j].v, (short)0, acc[i][j], false, false);
    }
#pragma unroll
    for (int i = 0; i < 2; ++i)
#pragma unroll
        for (int j = 0; j < 2; ++j)
#pragma unroll
            for (int g = 0; g < 8; ++g) {
                int row = m0 + i * 16 + g + 8 * hi;
                int col = n0 + j * 16 + ln;
                size_t o = (size_t)row * 512 + col;
                float v = acc[i][j][g];
                if (Rb) v += Rb[o];
                Cb[o] = v;
            }
}

// ---------------------------------------------------------------------------
// RoPE + head repack: y f32 [bc][h=512][w] -> out bf16 [bc][n][w][64]
// grid: total/256 with total = 32*8*512*32
// ---------------------------------------------------------------------------
__global__ void rope_repack_kernel(const float* __restrict__ y,
                                   unsigned short* __restrict__ out) {
    int idx = blockIdx.x * 256 + threadIdx.x;
    int w = idx & 511;
    int j = (idx >> 9) & 31;
    int n = (idx >> 14) & 7;
    int bc = idx >> 17;
    const float* yp = y + (size_t)bc * ELEMS;
    float t0 = yp[(size_t)(n * HD + 2 * j) * 512 + w];
    float t1 = yp[(size_t)(n * HD + 2 * j + 1) * 512 + w];
    float inv = __expf(-((2.f * j) * (1.f / HD)) * 9.210340371976184f); // ln(1e4)
    float ang = (float)w * inv;
    float cs = __cosf(ang), sn = __sinf(ang);
    float o0 = t0 * cs - t1 * sn;
    float o1 = t1 * cs + t0 * sn;
    unsigned short* qp = out + ((size_t)(bc * HEADS + n) * 512 + w) * HD + 2 * j;
    qp[0] = f32_to_bf16(o0);
    qp[1] = f32_to_bf16(o1);
}

// ---------------------------------------------------------------------------
// Flash-style attention per (bc, head): softmax(q kT / sqrt(512)) v
// q,k: bf16 [bc*8+n][w][64]; v: bf16 [bc][h][w] (== vT[d][u] per head)
// out aT: bf16 [bc][w][h]  (B-matrix layout for the wo GEMM)
// grid: BC*HEADS*8 blocks, block 128 (4 waves x 16 query rows)
// ---------------------------------------------------------------------------
__global__ __launch_bounds__(128) void attention_kernel(
    const unsigned short* __restrict__ qa,
    const unsigned short* __restrict__ ka,
    const unsigned short* __restrict__ vb,
    unsigned short* __restrict__ aT) {
    __shared__ __align__(16) unsigned short ldsP[4][16][32];
    int tid = threadIdx.x, lane = tid & 31, wid = tid >> 5;
    int ln = lane & 15, hi = lane >> 4;
    int rt = blockIdx.x & 7;
    int inst = blockIdx.x >> 3;     // bc*8 + n
    int bc = inst >> 3, n = inst & 7;
    int w0 = rt * 64 + wid * 16;

    const unsigned short* qp = qa + (size_t)inst * 512 * HD;
    const unsigned short* kp = ka + (size_t)inst * 512 * HD;
    const unsigned short* vp = vb + (size_t)bc * ELEMS + (size_t)n * HD * 512;
    const float scale = 0.044194173824159216f;   // 1/sqrt(512)

    Frag q0, q1;
    {
        const unsigned short* p = qp + (size_t)(w0 + ln) * HD + hi * 8;
        q0.q[0] = *(const uint4*)p;
        q0.q[1] = *(const uint4*)(p + 16);
        q1.q[0] = *(const uint4*)(p + 32);
        q1.q[1] = *(const uint4*)(p + 48);
    }
    float rmax[8], rsum[8];
    v8f oacc[4] = {};
#pragma unroll
    for (int g = 0; g < 8; ++g) { rmax[g] = -1e30f; rsum[g] = 0.f; }

    for (int u0 = 0; u0 < 512; u0 += 32) {
        v8f s[2] = {};
#pragma unroll
        for (int ut = 0; ut < 2; ++ut) {
            Frag b0, b1;
            const unsigned short* p = kp + (size_t)(u0 + ut * 16 + ln) * HD + hi * 8;
            b0.q[0] = *(const uint4*)p;        b0.q[1] = *(const uint4*)(p + 16);
            b1.q[0] = *(const uint4*)(p + 32); b1.q[1] = *(const uint4*)(p + 48);
            s[ut] = __builtin_amdgcn_wmma_f32_16x16x32_bf16(
                false, q0.v, false, b0.v, (short)0, s[ut], false, false);
            s[ut] = __builtin_amdgcn_wmma_f32_16x16x32_bf16(
                false, q1.v, false, b1.v, (short)0, s[ut], false, false);
        }
#pragma unroll
        for (int g = 0; g < 8; ++g) {
            float v0 = s[0][g] * scale, v1 = s[1][g] * scale;
            float lm = fmaxf(v0, v1);
            for (int m = 1; m < 16; m <<= 1) lm = fmaxf(lm, __shfl_xor(lm, m, 16));
            float mnew = fmaxf(rmax[g], lm);
            float f = __expf(rmax[g] - mnew);
            rmax[g] = mnew;
            float p0 = __expf(v0 - mnew), p1 = __expf(v1 - mnew);
            float ls = p0 + p1;
            for (int m = 1; m < 16; m <<= 1) ls += __shfl_xor(ls, m, 16);
            rsum[g] = rsum[g] * f + ls;
#pragma unroll
            for (int t = 0; t < 4; ++t) oacc[t][g] *= f;
            int row = g + 8 * hi;
            ldsP[wid][row][ln] = f32_to_bf16(p0);
            ldsP[wid][row][16 + ln] = f32_to_bf16(p1);
        }
        Frag pa;
        {
            const unsigned short* lp = &ldsP[wid][ln][hi * 8];
            pa.q[0] = *(const uint4*)lp;
            pa.q[1] = *(const uint4*)(lp + 16);
        }
#pragma unroll
        for (int dt = 0; dt < 4; ++dt) {
            Frag vf;
            const unsigned short* p = vp + (size_t)(dt * 16 + ln) * 512 + u0 + hi * 8;
            vf.q[0] = *(const uint4*)p;
            vf.q[1] = *(const uint4*)(p + 16);
            oacc[dt] = __builtin_amdgcn_wmma_f32_16x16x32_bf16(
                false, pa.v, false, vf.v, (short)0, oacc[dt], false, false);
        }
    }
    unsigned short* ap = aT + (size_t)bc * ELEMS;
#pragma unroll
    for (int dt = 0; dt < 4; ++dt)
#pragma unroll
        for (int g = 0; g < 8; ++g) {
            float val = oacc[dt][g] / rsum[g];
            int row = w0 + g + 8 * hi;
            int col = n * HD + dt * 16 + ln;
            ap[(size_t)row * 512 + col] = f32_to_bf16(val);
        }
}

// ---------------------------------------------------------------------------
// Channel mix 16->64 + SquaredReLU, output transposed bf16 [b][d][w][o]
// grid: B*512*512/256
// ---------------------------------------------------------------------------
__global__ __launch_bounds__(256) void mix1_kernel(
    const float* __restrict__ t1, const float* __restrict__ w1c,
    unsigned short* __restrict__ z1T) {
    __shared__ float wl[64 * 16];
    for (int i = threadIdx.x; i < 1024; i += 256) wl[i] = w1c[i];
    __syncthreads();
    int idx = blockIdx.x * 256 + threadIdx.x;
    int o = idx & 511;
    int w = (idx >> 9) & 511;
    int b = idx >> 18;
    float vin[16];
    const float* tp = t1 + (size_t)b * 16 * ELEMS + (size_t)o * 512 + w;
#pragma unroll
    for (int c = 0; c < 16; ++c) vin[c] = tp[(size_t)c * ELEMS];
    unsigned short* zp = z1T + (size_t)b * 64 * ELEMS + (size_t)w * 512 + o;
    for (int d = 0; d < 64; ++d) {
        float s = 0.f;
#pragma unroll
        for (int c = 0; c < 16; ++c) s += wl[d * 16 + c] * vin[c];
        s = fmaxf(s, 0.f);
        s = s * s;
        zp[(size_t)d * ELEMS] = f32_to_bf16(s);
    }
}

// ---------------------------------------------------------------------------
// Channel mix 64->16 + residual; per-batch. out[c][o][w] = h + sum w2c[c][cc]*t2
// grid: 512*512/256
// ---------------------------------------------------------------------------
__global__ __launch_bounds__(256) void mix2_kernel(
    const float* __restrict__ t2, const float* __restrict__ w2c,
    const float* __restrict__ h, float* __restrict__ out) {
    __shared__ float wl[16 * 64];
    for (int i = threadIdx.x; i < 1024; i += 256) wl[i] = w2c[i];
    __syncthreads();
    int idx = blockIdx.x * 256 + threadIdx.x;   // o*512 + w
    float vin[64];
    const float* tp = t2 + idx;
#pragma unroll
    for (int cc = 0; cc < 64; ++cc) vin[cc] = tp[(size_t)cc * ELEMS];
    for (int c = 0; c < 16; ++c) {
        float s = 0.f;
#pragma unroll
        for (int cc = 0; cc < 64; ++cc) s += wl[c * 64 + cc] * vin[cc];
        size_t oo = (size_t)c * ELEMS + idx;
        out[oo] = h[oo] + s;
    }
}

// ---------------------------------------------------------------------------
// Host launcher
// ---------------------------------------------------------------------------
extern "C" void kernel_launch(void* const* d_in, const int* in_sizes, int n_in,
                              void* d_out, int out_size, void* d_ws, size_t ws_size,
                              hipStream_t stream) {
    const float* x       = (const float*)d_in[0];
    const float* g1      = (const float*)d_in[1];
    const float* be1     = (const float*)d_in[2];
    const float* g2      = (const float*)d_in[3];
    const float* be2     = (const float*)d_in[4];
    const float* wq_conv = (const float*)d_in[5];
    const float* wq_lin  = (const float*)d_in[6];
    const float* wk_conv = (const float*)d_in[7];
    const float* wk_lin  = (const float*)d_in[8];
    const float* wv_conv = (const float*)d_in[9];
    const float* wv_lin  = (const float*)d_in[10];
    const float* wo_lin  = (const float*)d_in[11];
    const float* w1f     = (const float*)d_in[12];
    const float* w1c     = (const float*)d_in[13];
    const float* w2f     = (const float*)d_in[14];
    const float* w2c     = (const float*)d_in[15];
    float* out = (float*)d_out;

    char* ws = (char*)d_ws;
    // byte offsets (N = 8388608 elems)
    const size_t O_H    = 0;                        // f32 N (h residual)
    const size_t O_F32A = 33554432;                 // f32 N
    const size_t O_F32B = 67108864;                 // f32 N (A+B = t2 region)
    const size_t O_XT   = 100663296;                // bf16 N (transposed acts)
    const size_t O_WBF  = 117440512;                // bf16 up to 4N/... (w2f: 32MB)
    const size_t O_QA   = 150994944;                // bf16 N (also z1T region, 64MB)
    const size_t O_KA   = 167772160;                // bf16 N
    const size_t O_VB   = 184549376;                // bf16 N
    const size_t O_AT   = 201326592;                // bf16 N

    float* hbuf  = (float*)(ws + O_H);
    float* f32a  = (float*)(ws + O_F32A);
    float* f32b  = (float*)(ws + O_F32B);
    unsigned short* xT  = (unsigned short*)(ws + O_XT);
    unsigned short* wbf = (unsigned short*)(ws + O_WBF);
    unsigned short* qA  = (unsigned short*)(ws + O_QA);
    unsigned short* kA  = (unsigned short*)(ws + O_KA);
    unsigned short* vB  = (unsigned short*)(ws + O_VB);
    unsigned short* aT  = (unsigned short*)(ws + O_AT);
    unsigned short* z1T = qA;          // 64 MB region (QA..AT+16MB), attention done
    float* t2 = f32a;                  // 64 MB region (F32A+F32B), per-batch

    const int NTOT = HW_B * HW_C * ELEMS;         // 8388608
    dim3 trGrid(16, 16, 32), trBlk(32, 8);

    // ---- LN1 ----
    layernorm_kernel<<<dim3(128), 128, 0, stream>>>(x, g1, be1, f32a);

    // ---- q / k / v paths ----
    const float* convW[3] = {wq_conv, wk_conv, wv_conv};
    const float* linW[3]  = {wq_lin, wk_lin, wv_lin};
    for (int p = 0; p < 3; ++p) {
        conv3x3_kernel<<<NTOT / 256, 256, 0, stream>>>(f32a, convW[p], f32b);
        transpose_cvt_kernel<<<trGrid, trBlk, 0, stream>>>(f32b, xT);
        cvt_bf16_kernel<<<(16 * ELEMS) / 256, 256, 0, stream>>>(linW[p], wbf, 16 * ELEMS);
        gemm_bf16_kernel<<<dim3(8, 4, 32), 256, 0, stream>>>(wbf, 16, xT, nullptr, f32b);
        if (p == 0)
            rope_repack_kernel<<<(32 * 8 * 512 * 32) / 256, 256, 0, stream>>>(f32b, qA);
        else if (p == 1)
            rope_repack_kernel<<<(32 * 8 * 512 * 32) / 256, 256, 0, stream>>>(f32b, kA);
        else
            cvt_bf16_kernel<<<NTOT / 256, 256, 0, stream>>>(f32b, vB, NTOT);
    }

    // ---- attention ----
    attention_kernel<<<2048, 128, 0, stream>>>(qA, kA, vB, aT);

    // ---- output projection + residual: h = x + a @ wo ----
    cvt_bf16_kernel<<<(16 * ELEMS) / 256, 256, 0, stream>>>(wo_lin, wbf, 16 * ELEMS);
    gemm_bf16_kernel<<<dim3(8, 4, 32), 256, 0, stream>>>(wbf, 16, aT, x, hbuf);

    // ---- FFN ----
    layernorm_kernel<<<dim3(128), 128, 0, stream>>>(hbuf, g2, be2, f32a);
    transpose_cvt_kernel<<<trGrid, trBlk, 0, stream>>>(f32a, xT);
    cvt_bf16_kernel<<<(16 * ELEMS) / 256, 256, 0, stream>>>(w1f, wbf, 16 * ELEMS);
    gemm_bf16_kernel<<<dim3(8, 4, 32), 256, 0, stream>>>(wbf, 16, xT, nullptr, f32b);
    mix1_kernel<<<(HW_B * ELEMS) / 256, 256, 0, stream>>>(f32b, w1c, z1T);
    cvt_bf16_kernel<<<(64 * ELEMS) / 256, 256, 0, stream>>>(w2f, wbf, 64 * ELEMS);
    for (int b = 0; b < HW_B; ++b) {
        gemm_bf16_kernel<<<dim3(8, 4, 64), 256, 0, stream>>>(
            wbf, 64, z1T + (size_t)b * 64 * ELEMS, nullptr, t2);
        mix2_kernel<<<ELEMS / 256, 256, 0, stream>>>(
            t2, w2c, hbuf + (size_t)b * 16 * ELEMS, out + (size_t)b * 16 * ELEMS);
    }
    (void)in_sizes; (void)n_in; (void)out_size; (void)ws_size;
}
